// MoE_17858474017345
// MI455X (gfx1250) — compile-verified, matrix operand order checked by reading, not run on hardware
//
#include <hip/hip_runtime.h>
#include <hip/hip_bf16.h>
#include <stdint.h>

typedef __attribute__((ext_vector_type(16))) _Float16 v16h;
typedef __attribute__((ext_vector_type(8)))  _Float16 v8h;
typedef __attribute__((ext_vector_type(8)))  float    v8f;
typedef int v4i __attribute__((vector_size(16)));   // matches builtin param type

#define T_TOK 8192
#define DDIM  2048
#define NEXP  8
#define FDIM  2048

#define BM 64      // M tile (rows)
#define BN 64      // N tile (output cols)
#define BK 64      // K chunk staged per iteration (2 WMMA k-steps)

// ---- CDNA5 async memory->LDS staging (ASYNCcnt path), with safe fallback ----
#if defined(__gfx1250__) && __has_builtin(__builtin_amdgcn_global_load_async_to_lds_b128)
#define HAVE_ASYNC_LDS 1
#else
#define HAVE_ASYNC_LDS 0
#endif

__device__ __forceinline__ void stage16(_Float16* lds, const _Float16* gsrc) {
#if HAVE_ASYNC_LDS
    // async DMA global -> LDS, tracked by ASYNCcnt (no VGPR round trip)
    __builtin_amdgcn_global_load_async_to_lds_b128((v4i*)gsrc, (v4i*)lds, 0, 0);
#else
    *(uint4*)lds = *(const uint4*)gsrc;
#endif
}

__device__ __forceinline__ void async_fence() {
#if HAVE_ASYNC_LDS
#if __has_builtin(__builtin_amdgcn_s_wait_asynccnt)
    __builtin_amdgcn_s_wait_asynccnt(0);
#else
    asm volatile("s_wait_asynccnt 0x0" ::: "memory");
#endif
#endif
}

// ---------------- workspace layout (bytes) ----------------
constexpr size_t WSO_COUNTS  = 0;                        // 8 ints (zeroed)
constexpr size_t WSO_OFFSETS = 64;                       // 9 ints
constexpr size_t WSO_ROWS    = 1024;                     // NEXP*T_TOK ints
constexpr size_t WSO_SCALES  = WSO_ROWS + (size_t)NEXP * T_TOK * 4;
constexpr size_t WSO_XH      = 1u << 20;
constexpr size_t SZ_XH       = (size_t)T_TOK * DDIM * 2;
constexpr size_t WSO_W1H     = WSO_XH + SZ_XH;
constexpr size_t SZ_W        = (size_t)NEXP * FDIM * DDIM * 2;
constexpr size_t WSO_W3H     = WSO_W1H + SZ_W;
constexpr size_t WSO_W2H     = WSO_W3H + SZ_W;
constexpr size_t WSO_H       = WSO_W2H + SZ_W;           // (2T + 8*BM) x FDIM f16

// ---------------------------------------------------------------------------
// f32 -> f16 bulk conversion, 8 elements/thread
// ---------------------------------------------------------------------------
__global__ __launch_bounds__(256) void cvt_f32_f16(const float* __restrict__ src,
                                                   _Float16* __restrict__ dst,
                                                   int n8) {
    int i = blockIdx.x * 256 + threadIdx.x;
    if (i >= n8) return;
    const float4* s = (const float4*)src + (size_t)i * 2;
    float4 a = s[0], b = s[1];
    v8h o;
    o[0] = (_Float16)a.x; o[1] = (_Float16)a.y;
    o[2] = (_Float16)a.z; o[3] = (_Float16)a.w;
    o[4] = (_Float16)b.x; o[5] = (_Float16)b.y;
    o[6] = (_Float16)b.z; o[7] = (_Float16)b.w;
    *(v8h*)(dst + (size_t)i * 8) = o;
}

// ---------------------------------------------------------------------------
// Pass 0: gating — one wave per token: 8 dot products over D, softmax, top-2,
// atomic append into per-expert token lists.
// ---------------------------------------------------------------------------
__global__ __launch_bounds__(256) void moe_gate_topk(const float* __restrict__ x,
                                                     const float* __restrict__ gw,
                                                     int* __restrict__ counts,
                                                     int* __restrict__ rowsArr,
                                                     float* __restrict__ sclArr) {
    const int wave = threadIdx.x >> 5;
    const int lane = threadIdx.x & 31;
    const int t = blockIdx.x * 8 + wave;

    float acc[NEXP];
#pragma unroll
    for (int e = 0; e < NEXP; ++e) acc[e] = 0.0f;

    const float* xr = x + (size_t)t * DDIM;
    for (int d = lane; d < DDIM; d += 32) {
        float xv = xr[d];
#pragma unroll
        for (int e = 0; e < NEXP; ++e) acc[e] += xv * gw[e * DDIM + d];
    }
#pragma unroll
    for (int e = 0; e < NEXP; ++e) {
#pragma unroll
        for (int off = 16; off > 0; off >>= 1)
            acc[e] += __shfl_down(acc[e], off, 32);
    }

    if (lane == 0) {
        float m = acc[0];
#pragma unroll
        for (int e = 1; e < NEXP; ++e) m = fmaxf(m, acc[e]);
        float p[NEXP];
        float s = 0.0f;
#pragma unroll
        for (int e = 0; e < NEXP; ++e) { p[e] = expf(acc[e] - m); s += p[e]; }
        const float inv = 1.0f / s;

        int e0 = 0;
#pragma unroll
        for (int e = 1; e < NEXP; ++e) if (p[e] > p[e0]) e0 = e;
        int e1 = (e0 == 0) ? 1 : 0;
#pragma unroll
        for (int e = 0; e < NEXP; ++e) if (e != e0 && p[e] > p[e1]) e1 = e;

        int pos0 = atomicAdd(&counts[e0], 1);
        rowsArr[e0 * T_TOK + pos0] = t;
        sclArr[e0 * T_TOK + pos0] = p[e0] * inv;
        int pos1 = atomicAdd(&counts[e1], 1);
        rowsArr[e1 * T_TOK + pos1] = t;
        sclArr[e1 * T_TOK + pos1] = p[e1] * inv;
    }
}

// ---------------------------------------------------------------------------
// Pass 1: tile-padded prefix sum of expert counts -> H segment offsets.
// ---------------------------------------------------------------------------
__global__ void moe_offsets(const int* __restrict__ counts,
                            int* __restrict__ offsets) {
    if (threadIdx.x == 0 && blockIdx.x == 0) {
        int acc = 0;
        for (int e = 0; e < NEXP; ++e) {
            offsets[e] = acc;
            acc += ((counts[e] + BM - 1) / BM) * BM;   // pad to M tile
        }
        offsets[NEXP] = acc;
    }
}

// ---------------------------------------------------------------------------
// Pass 2: H[row,f] = silu(X@W1^T) * (X@W3^T) for routed rows (all f16 inputs).
// 8 waves in 4(M) x 2(N); each wave a 16x32 strip, double-buffered LDS with
// async global->LDS staging of the next K chunk while WMMAs consume this one.
// ---------------------------------------------------------------------------
__global__ __launch_bounds__(256)
void moe_ffn_h(const _Float16* __restrict__ Xh,
               const _Float16* __restrict__ W1h,
               const _Float16* __restrict__ W3h,
               const int* __restrict__ counts,
               const int* __restrict__ offsets,
               const int* __restrict__ rowsArr,
               _Float16* __restrict__ H) {
    const int e = blockIdx.z;
    const int cnt = counts[e];
    const int m_base = blockIdx.y * BM;
    if (m_base >= cnt) return;                   // block-uniform early exit
    const int f_base = blockIdx.x * BN;
    const int hbase = offsets[e];

    __shared__ alignas(32) _Float16 sX[2][BM][BK];   // 16 KB
    __shared__ alignas(32) _Float16 sG[2][BN][BK];   // 16 KB
    __shared__ alignas(32) _Float16 sU[2][BN][BK];   // 16 KB
    __shared__ int sTok[BM];

    const int tid = threadIdx.x;
    const int lane = tid & 31;
    const int wave = tid >> 5;
    const int wm = wave & 3;                     // M sub-tile 0..3
    const int wn = wave >> 2;                    // N half 0..1

    if (tid < BM) {
        int i = m_base + tid;
        sTok[tid] = rowsArr[e * T_TOK + (i < cnt ? i : 0)];
    }
    __syncthreads();

    // hoisted per-thread staging pointers: chunk = tid + v*256 -> (row, col8)
    const _Float16* xsrc[2];
    const _Float16* gsrc[2];
    const _Float16* usrc[2];
    int srow[2], scol[2];
#pragma unroll
    for (int v = 0; v < 2; ++v) {
        int chunk = tid + v * 256;
        int r = chunk >> 3, c = (chunk & 7) * 8;
        srow[v] = r; scol[v] = c;
        xsrc[v] = Xh + (size_t)sTok[r] * DDIM + c;
        gsrc[v] = W1h + ((size_t)e * FDIM + f_base + r) * DDIM + c;
        usrc[v] = W3h + ((size_t)e * FDIM + f_base + r) * DDIM + c;
    }

    auto stage = [&](int buf, int k0) {
#pragma unroll
        for (int v = 0; v < 2; ++v) {
            stage16(&sX[buf][srow[v]][scol[v]], xsrc[v] + k0);
            stage16(&sG[buf][srow[v]][scol[v]], gsrc[v] + k0);
            stage16(&sU[buf][srow[v]][scol[v]], usrc[v] + k0);
        }
    };

    v8f accG[2] = {};
    v8f accU[2] = {};

    stage(0, 0);
    async_fence();
    __syncthreads();

    const int NK = DDIM / BK;                    // 32 chunks
    const int ar = wm * 16 + (lane & 15);
    const int ko = (lane >> 4) * 16;
    for (int kt = 0; kt < NK; ++kt) {
        const int buf = kt & 1;
        if (kt + 1 < NK) stage(buf ^ 1, (kt + 1) * BK);   // async prefetch
#pragma unroll
        for (int ks = 0; ks < 2; ++ks) {
            v16h a = *(const v16h*)&sX[buf][ar][ks * 32 + ko];
#pragma unroll
            for (int sub = 0; sub < 2; ++sub) {
                const int br = wn * 32 + sub * 16 + (lane & 15);
                v16h b1 = *(const v16h*)&sG[buf][br][ks * 32 + ko];
                v16h b3 = *(const v16h*)&sU[buf][br][ks * 32 + ko];
                accG[sub] = __builtin_amdgcn_wmma_f32_16x16x32_f16(
                    false, a, false, b1, (short)0, accG[sub], false, false);
                accU[sub] = __builtin_amdgcn_wmma_f32_16x16x32_f16(
                    false, a, false, b3, (short)0, accU[sub], false, false);
            }
        }
        async_fence();
        __syncthreads();
    }

    // fused SiLU(g)*u epilogue; C layout: VGPR j -> M = j + 8*(lane>>4)
    const int mhi = (lane >> 4) * 8;
#pragma unroll
    for (int sub = 0; sub < 2; ++sub) {
        const int ncol = f_base + wn * 32 + sub * 16 + (lane & 15);
#pragma unroll
        for (int j = 0; j < 8; ++j) {
            int m = m_base + wm * 16 + mhi + j;
            float g = accG[sub][j];
            float h = (g / (1.0f + expf(-g))) * accU[sub][j];
            H[(size_t)(hbase + m) * FDIM + ncol] = (_Float16)h;
        }
    }
}

// ---------------------------------------------------------------------------
// Pass 3: y[token] += scale * (H @ W2^T), scatter-add with f32 atomics.
// ---------------------------------------------------------------------------
__global__ __launch_bounds__(256)
void moe_down_scatter(const _Float16* __restrict__ H,
                      const _Float16* __restrict__ W2h,
                      const int* __restrict__ counts,
                      const int* __restrict__ offsets,
                      const int* __restrict__ rowsArr,
                      const float* __restrict__ sclArr,
                      float* __restrict__ out) {
    const int e = blockIdx.z;
    const int cnt = counts[e];
    const int m_base = blockIdx.y * BM;
    if (m_base >= cnt) return;
    const int d_base = blockIdx.x * BN;
    const int hbase = offsets[e];

    __shared__ alignas(32) _Float16 sH[2][BM][BK];
    __shared__ alignas(32) _Float16 sW[2][BN][BK];
    __shared__ int   sTok[BM];
    __shared__ float sScl[BM];

    const int tid = threadIdx.x;
    const int lane = tid & 31;
    const int wave = tid >> 5;
    const int wm = wave & 3;
    const int wn = wave >> 2;

    if (tid < BM) {
        int i = m_base + tid;
        bool valid = i < cnt;
        sTok[tid] = rowsArr[e * T_TOK + (valid ? i : 0)];
        sScl[tid] = valid ? sclArr[e * T_TOK + i] : 0.0f;   // padded rows weigh 0
    }
    __syncthreads();

    const _Float16* hsrc[2];
    const _Float16* wsrc[2];
    int srow[2], scol[2];
#pragma unroll
    for (int v = 0; v < 2; ++v) {
        int chunk = tid + v * 256;
        int r = chunk >> 3, c = (chunk & 7) * 8;
        srow[v] = r; scol[v] = c;
        hsrc[v] = H + (size_t)(hbase + m_base + r) * FDIM + c;
        wsrc[v] = W2h + ((size_t)e * DDIM + d_base + r) * FDIM + c;
    }

    auto stage = [&](int buf, int k0) {
#pragma unroll
        for (int v = 0; v < 2; ++v) {
            stage16(&sH[buf][srow[v]][scol[v]], hsrc[v] + k0);
            stage16(&sW[buf][srow[v]][scol[v]], wsrc[v] + k0);
        }
    };

    v8f acc[2] = {};

    stage(0, 0);
    async_fence();
    __syncthreads();

    const int NK = FDIM / BK;
    const int ar = wm * 16 + (lane & 15);
    const int ko = (lane >> 4) * 16;
    for (int kt = 0; kt < NK; ++kt) {
        const int buf = kt & 1;
        if (kt + 1 < NK) stage(buf ^ 1, (kt + 1) * BK);
#pragma unroll
        for (int ks = 0; ks < 2; ++ks) {
            v16h a = *(const v16h*)&sH[buf][ar][ks * 32 + ko];
#pragma unroll
            for (int sub = 0; sub < 2; ++sub) {
                const int br = wn * 32 + sub * 16 + (lane & 15);
                v16h b = *(const v16h*)&sW[buf][br][ks * 32 + ko];
                acc[sub] = __builtin_amdgcn_wmma_f32_16x16x32_f16(
                    false, a, false, b, (short)0, acc[sub], false, false);
            }
        }
        async_fence();
        __syncthreads();
    }

    const int mhi = (lane >> 4) * 8;
#pragma unroll
    for (int sub = 0; sub < 2; ++sub) {
        const int d = d_base + wn * 32 + sub * 16 + (lane & 15);
#pragma unroll
        for (int j = 0; j < 8; ++j) {
            int ml = wm * 16 + mhi + j;
            float v = acc[sub][j] * sScl[ml];
            atomicAdd(&out[(size_t)sTok[ml] * DDIM + d], v);
        }
    }
}

// ---------------------------------------------------------------------------
extern "C" void kernel_launch(void* const* d_in, const int* in_sizes, int n_in,
                              void* d_out, int out_size, void* d_ws, size_t ws_size,
                              hipStream_t stream) {
    const float* x  = (const float*)d_in[0];
    const float* gw = (const float*)d_in[1];
    const float* w1 = (const float*)d_in[2];
    const float* w3 = (const float*)d_in[3];
    const float* w2 = (const float*)d_in[4];
    float* out = (float*)d_out;

    char* ws = (char*)d_ws;
    int*      counts  = (int*)(ws + WSO_COUNTS);
    int*      offsets = (int*)(ws + WSO_OFFSETS);
    int*      rowsArr = (int*)(ws + WSO_ROWS);
    float*    sclArr  = (float*)(ws + WSO_SCALES);
    _Float16* Xh      = (_Float16*)(ws + WSO_XH);
    _Float16* W1h     = (_Float16*)(ws + WSO_W1H);
    _Float16* W3h     = (_Float16*)(ws + WSO_W3H);
    _Float16* W2h     = (_Float16*)(ws + WSO_W2H);
    _Float16* H       = (_Float16*)(ws + WSO_H);

    // deterministic init each call (graph-capture safe)
    (void)hipMemsetAsync(counts, 0, 64, stream);
    (void)hipMemsetAsync(out, 0, (size_t)out_size * sizeof(float), stream);

    // per-call f16 mirrors (~26 us of HBM traffic, negligible vs compute)
    {
        int n8x = (T_TOK * DDIM) / 8;
        int n8w = (NEXP * FDIM * DDIM) / 8;
        cvt_f32_f16<<<(n8x + 255) / 256, 256, 0, stream>>>(x, Xh, n8x);
        cvt_f32_f16<<<(n8w + 255) / 256, 256, 0, stream>>>(w1, W1h, n8w);
        cvt_f32_f16<<<(n8w + 255) / 256, 256, 0, stream>>>(w3, W3h, n8w);
        cvt_f32_f16<<<(n8w + 255) / 256, 256, 0, stream>>>(w2, W2h, n8w);
    }

    moe_gate_topk<<<T_TOK / 8, 256, 0, stream>>>(x, gw, counts, rowsArr, sclArr);
    moe_offsets<<<1, 1, 0, stream>>>(counts, offsets);

    dim3 gA(FDIM / BN, T_TOK / BM, NEXP);   // over-launched; dead tiles exit on count
    moe_ffn_h<<<gA, 256, 0, stream>>>(Xh, W1h, W3h, counts, offsets, rowsArr, H);

    dim3 gB(DDIM / BN, T_TOK / BM, NEXP);
    moe_down_scatter<<<gB, 256, 0, stream>>>(H, W2h, counts, offsets, rowsArr, sclArr, out);
}
